// MV_LOCAL_ATTN_87806311399751
// MI455X (gfx1250) — matrix operand order repaired and checked by reading, compiled, not run on hardware
//
#include <hip/hip_runtime.h>

typedef _Float16 half_t;
typedef __attribute__((ext_vector_type(16))) _Float16 v16h;
typedef __attribute__((ext_vector_type(8)))  _Float16 v8h;
typedef __attribute__((ext_vector_type(8)))  float    v8f;

constexpr int Bc = 4, Cc = 64, Hh = 128, Ww = 128;
constexpr int K2 = 9, NF = 64, CIN = 1152;
constexpr int WAVES = 4, PX = 16;
constexpr int W1P_HALVES = 36 * 4 * 512;   // 36 K-chunks x 4 N-tiles x (32 lanes x 16 halves)
constexpr int W2P_HALVES = 2 * 512;        // 2 K-chunks x 1 N-tile
constexpr float MV_SCALE = 32.0f;

// ---------------------------------------------------------------------------
// Pre-swizzle w1/w2 (f32 -> f16) into WMMA B-fragment order.
// Our GEMM K order: K in [0,576) = cen patches (tap-major: K = tap*64 + c),
//                   K in [576,1152) = aligned patches (same tap-major order).
// Original w1 column order: [aligned c*9+tap (576) | cen c*9+tap (576)].
// B-fragment order used here (f16 32x16): lane l holds col n = l&15, rows
// K = (l>>4)*16 + e for e=0..15. We store lane-contiguous: tile*512 + l*16 + e.
// ---------------------------------------------------------------------------
__global__ void repack_weights(const float* __restrict__ w1,
                               const float* __restrict__ w2,
                               half_t* __restrict__ w1p,
                               half_t* __restrict__ w2p) {
  int idx = blockIdx.x * blockDim.x + threadIdx.x;
  if (idx < W1P_HALVES) {
    int t  = idx >> 9;         // tile = kc*4 + nt
    int l  = (idx >> 4) & 31;  // lane
    int e  = idx & 15;
    int kc = t >> 2, nt = t & 3;
    int K  = kc * 32 + ((l >> 4) * 16) + e;
    int n  = nt * 16 + (l & 15);
    int origCol;
    if (K < 576) {                       // cen half
      int k9 = K >> 6, c = K & 63;
      origCol = 576 + c * 9 + k9;
    } else {                             // aligned half
      int Kp = K - 576;
      int k9 = Kp >> 6, c = Kp & 63;
      origCol = c * 9 + k9;
    }
    w1p[idx] = (half_t)w1[n * CIN + origCol];
  } else if (idx < W1P_HALVES + W2P_HALVES) {
    int j = idx - W1P_HALVES;
    int t = j >> 9;
    int l = (j >> 4) & 31;
    int e = j & 15;
    int K = t * 32 + ((l >> 4) * 16) + e;   // 0..63 over nf
    int n = l & 15;                          // 9 valid outputs, rest zero
    w2p[j] = (n < K2) ? (half_t)w2[n * NF + K] : (half_t)0.0f;
  }
}

// Build a 16-half fragment from two 16-byte LDS/global loads.
// gap = 16 -> halves [base..base+7] and [base+16..base+23] (unused)
// gap = 8  -> A- and B-fragments (16 contiguous halves)
__device__ inline v16h ld16(const half_t* p, int gap) {
  v8h lo = *(const v8h*)p;
  v8h hi = *(const v8h*)(p + gap);
  return __builtin_shufflevector(lo, hi, 0, 1, 2, 3, 4, 5, 6, 7,
                                         8, 9, 10, 11, 12, 13, 14, 15);
}

// ---------------------------------------------------------------------------
// Fused kernel: one wave = 16 consecutive pixels of one row.
// Phase A1: stage cen 3x3 patches (f16) in LDS   -> Phase B1: 18 K-chunks GEMM1
// Phase A2: stage warped 3x3 patches (f16)       -> Phase B2: 18 K-chunks GEMM1
// bias+LeakyReLU -> GEMM2 (64->9) -> softmax(9) -> attn-weighted mean -> out
// ---------------------------------------------------------------------------
__global__ __launch_bounds__(WAVES * 32) void mv_local_attn_kernel(
    const float* __restrict__ nbh, const float* __restrict__ cen,
    const float* __restrict__ mv,
    const half_t* __restrict__ w1p, const float* __restrict__ b1,
    const half_t* __restrict__ w2p, const float* __restrict__ b2,
    float* __restrict__ out) {
  extern __shared__ char smemRaw[];
  half_t* stage = (half_t*)smemRaw;             // WAVES * 16px * 576 halves
  half_t* h1s   = stage + WAVES * PX * 576;     // WAVES * 16px * 64 halves
  float*  attn  = (float*)(h1s + WAVES * PX * 64); // WAVES * 16px * 16 floats

  const int tid  = threadIdx.x;
  const int wave = tid >> 5;
  const int lane = tid & 31;
  const int px   = lane & 15;   // pixel within wave tile (== WMMA row for A)
  const int ch   = lane >> 4;   // which 32-channel half this lane stages

  const int pixelBase = blockIdx.x * (WAVES * PX) + wave * PX;
  const int b   = pixelBase >> 14;        // / (H*W)
  const int rem = pixelBase & 16383;
  const int y   = rem >> 7;
  const int x   = (rem & 127) + px;       // tiles are 16-aligned in x, never wrap

  // --- per-pixel warp coordinates (grid_sample, border padding, align_corners)
  const float u  = mv[(((size_t)b * 2 + 0) * Hh + y) * Ww + x] * (2.0f * MV_SCALE / Ww);
  const float v  = mv[(((size_t)b * 2 + 1) * Hh + y) * Ww + x] * (2.0f * MV_SCALE / Hh);
  const float xv = -1.0f + 2.0f * x / (float)(Ww - 1);
  const float yv = -1.0f + 2.0f * y / (float)(Hh - 1);
  const float gx = fminf(fmaxf(xv + u, -1.0f), 1.0f);
  const float gy = fminf(fmaxf(yv + v, -1.0f), 1.0f);
  const float xs = (gx + 1.0f) * 0.5f * (Ww - 1);
  const float ys = (gy + 1.0f) * 0.5f * (Hh - 1);
  const float x0f = floorf(xs), y0f = floorf(ys);
  const float wx = xs - x0f, wy = ys - y0f;
  const int x0i = min(max((int)x0f, 0), Ww - 1);
  const int y0i = min(max((int)y0f, 0), Hh - 1);
  const int dx = min(x0i + 1, Ww - 1) - x0i;
  const int dy = min(y0i + 1, Hh - 1) - y0i;

  half_t* myStage = stage + wave * PX * 576;
  const int cbase = ch * 32;

  // ---- phase A1: central 3x3 zero-padded patches -> LDS (f16) ----
  for (int ci = 0; ci < 32; ++ci) {
    const int c = cbase + ci;
    const float* p = cen + (size_t)(b * Cc + c) * Hh * Ww;
    #pragma unroll
    for (int i = 0; i < 3; ++i) {
      #pragma unroll
      for (int j = 0; j < 3; ++j) {
        const int rr = y - 1 + i, cc2 = x - 1 + j;
        const bool ok = (rr >= 0) & (rr < Hh) & (cc2 >= 0) & (cc2 < Ww);
        const float val = ok ? p[rr * Ww + cc2] : 0.0f;
        myStage[px * 576 + (i * 3 + j) * 64 + c] = (half_t)val;
      }
    }
  }
  __syncthreads();

  // ---- phase B1: GEMM1 over cen half of K (chunks 0..17) ----
  v8f acc[4];
  #pragma unroll
  for (int nt = 0; nt < 4; ++nt) acc[nt] = (v8f)(0.0f);

  for (int kc = 0; kc < 18; ++kc) {
    v16h a = ld16(myStage + px * 576 + kc * 32 + ch * 16, 8);
    #pragma unroll
    for (int nt = 0; nt < 4; ++nt) {
      v16h bb = ld16(w1p + (size_t)(kc * 4 + nt) * 512 + lane * 16, 8);
      acc[nt] = __builtin_amdgcn_wmma_f32_16x16x32_f16(
          false, a, false, bb, (short)0, acc[nt], false, false);
    }
  }
  __syncthreads();

  // ---- phase A2: motion-aligned patches. All 9 taps of one channel come
  //      from one 4x4 neighborhood (zero-padded unfold + border-clamped
  //      bilinear share x0i/y0i/dx/dy/wx/wy). Aligned data stays resident
  //      in LDS for the final reduction. ----
  const float omx = 1.0f - wx, omy = 1.0f - wy;
  for (int ci = 0; ci < 32; ++ci) {
    const int c = cbase + ci;
    const float* p = nbh + (size_t)(b * Cc + c) * Hh * Ww;
    float vv[4][4];
    #pragma unroll
    for (int r = 0; r < 4; ++r) {
      const int rr = y0i - 1 + r;
      #pragma unroll
      for (int s = 0; s < 4; ++s) {
        const int cc2 = x0i - 1 + s;
        const bool ok = (rr >= 0) & (rr < Hh) & (cc2 >= 0) & (cc2 < Ww);
        vv[r][s] = ok ? p[rr * Ww + cc2] : 0.0f;
      }
    }
    #pragma unroll
    for (int i = 0; i < 3; ++i) {
      #pragma unroll
      for (int j = 0; j < 3; ++j) {
        const float top = omx * vv[i][j]      + wx * vv[i][j + dx];
        const float bot = omx * vv[i + dy][j] + wx * vv[i + dy][j + dx];
        myStage[px * 576 + (i * 3 + j) * 64 + c] = (half_t)(omy * top + wy * bot);
      }
    }
  }
  __syncthreads();

  // ---- phase B2: GEMM1 over aligned half of K (chunks 18..35) ----
  for (int kc = 18; kc < 36; ++kc) {
    v16h a = ld16(myStage + px * 576 + (kc - 18) * 32 + ch * 16, 8);
    #pragma unroll
    for (int nt = 0; nt < 4; ++nt) {
      v16h bb = ld16(w1p + (size_t)(kc * 4 + nt) * 512 + lane * 16, 8);
      acc[nt] = __builtin_amdgcn_wmma_f32_16x16x32_f16(
          false, a, false, bb, (short)0, acc[nt], false, false);
    }
  }

  // ---- bias + LeakyReLU(0.1); stash h1 (f16) for GEMM2 (A-layout via LDS) ----
  // C/D layout: acc[nt][r] is (M = r + 8*(lane>=16), N = nt*16 + (lane&15))
  #pragma unroll
  for (int nt = 0; nt < 4; ++nt) {
    const float bias = b1[nt * 16 + (lane & 15)];
    #pragma unroll
    for (int r = 0; r < 8; ++r) {
      float hv = acc[nt][r] + bias;
      hv = (hv >= 0.0f) ? hv : 0.1f * hv;
      const int m = r + ch * 8;
      h1s[wave * PX * 64 + m * 64 + nt * 16 + (lane & 15)] = (half_t)hv;
    }
  }
  __syncthreads();

  // ---- GEMM2: h2 = h1(16x64) x w2^T (9 valid cols of N=16) ----
  v8f acc2 = (v8f)(0.0f);
  #pragma unroll
  for (int kc = 0; kc < 2; ++kc) {
    v16h a  = ld16(h1s + wave * PX * 64 + px * 64 + kc * 32 + ch * 16, 8);
    v16h bb = ld16(w2p + kc * 512 + lane * 16, 8);
    acc2 = __builtin_amdgcn_wmma_f32_16x16x32_f16(
        false, a, false, bb, (short)0, acc2, false, false);
  }
  const float bias2 = ((lane & 15) < K2) ? b2[lane & 15] : 0.0f;
  #pragma unroll
  for (int r = 0; r < 8; ++r) {
    const int m = r + ch * 8;
    attn[wave * PX * 16 + m * 16 + (lane & 15)] = acc2[r] + bias2;
  }
  __syncthreads();

  // ---- softmax over 9 taps: lanes 0..15 each own one pixel ----
  if (lane < 16) {
    float* ap = attn + wave * PX * 16 + lane * 16;
    float mx = ap[0];
    #pragma unroll
    for (int k = 1; k < K2; ++k) mx = fmaxf(mx, ap[k]);
    float s = 0.0f, e[K2];
    #pragma unroll
    for (int k = 0; k < K2; ++k) { e[k] = __expf(ap[k] - mx); s += e[k]; }
    const float inv = 1.0f / s;
    #pragma unroll
    for (int k = 0; k < K2; ++k) ap[k] = e[k] * inv;
  }
  __syncthreads();

  // ---- output: mean over taps of aligned * attn (aligned still in LDS) ----
  const float* ap = attn + wave * PX * 16 + px * 16;
  for (int ci = 0; ci < 32; ++ci) {
    const int c = cbase + ci;
    float s = 0.0f;
    #pragma unroll
    for (int k = 0; k < K2; ++k)
      s += (float)myStage[px * 576 + k * 64 + c] * ap[k];
    out[((size_t)(b * Cc + c) * Hh + y) * Ww + x] = s * (1.0f / 9.0f);
  }
}

extern "C" void kernel_launch(void* const* d_in, const int* in_sizes, int n_in,
                              void* d_out, int out_size, void* d_ws, size_t ws_size,
                              hipStream_t stream) {
  const float* nbh = (const float*)d_in[0];
  const float* cen = (const float*)d_in[1];
  const float* mv  = (const float*)d_in[2];
  const float* w1  = (const float*)d_in[3];
  const float* b1  = (const float*)d_in[4];
  const float* w2  = (const float*)d_in[5];
  const float* b2  = (const float*)d_in[6];
  float* out = (float*)d_out;

  half_t* w1p = (half_t*)d_ws;                // 147456 B
  half_t* w2p = w1p + W1P_HALVES;             // + 2048 B

  {
    const int n = W1P_HALVES + W2P_HALVES;
    repack_weights<<<(n + 255) / 256, 256, 0, stream>>>(w1, w2, w1p, w2p);
  }

  const int smem = WAVES * PX * 576 * 2   // fuse stage (f16)
                 + WAVES * PX * 64 * 2    // h1 stash (f16)
                 + WAVES * PX * 16 * 4;   // attention logits (f32)  = 86016 B
  hipFuncSetAttribute(reinterpret_cast<const void*>(mv_local_attn_kernel),
                      hipFuncAttributeMaxDynamicSharedMemorySize, smem);

  const int totalPixels = Bc * Hh * Ww;
  const int blocks = totalPixels / (WAVES * PX);   // 1024
  mv_local_attn_kernel<<<blocks, WAVES * 32, smem, stream>>>(
      nbh, cen, mv, w1p, b1, w2p, b2, out);
}
